// MultiheadedAttention_1726576853828
// MI455X (gfx1250) — matrix operand order repaired and checked by reading, not compiled, not run on hardware
//
#include <hip/hip_runtime.h>
#include <hip/hip_bf16.h>

typedef __attribute__((ext_vector_type(16))) _Float16 v16h;
typedef __attribute__((ext_vector_type(8)))  _Float16 v8h;
typedef __attribute__((ext_vector_type(8)))  float    v8f;

#define WMMA_F16F32(a, b, c) \
    __builtin_amdgcn_wmma_f32_16x16x32_f16(false, (a), false, (b), (short)0, (c), false, false)

static constexpr int B = 4, T = 1024, H = 16, S = 64, E = 1024;
static constexpr float NEGV = -1000000000.0f;
static constexpr float QK_SCALE = 1.0f / 32.0f;   // (e^0.25)^-2 with e=1024

__device__ __forceinline__ v8f zero8f() {
    v8f z;
#pragma unroll
    for (int i = 0; i < 8; ++i) z[i] = 0.0f;
    return z;
}

__device__ __forceinline__ v16h cat8(v8h lo, v8h hi) {
    v16h r;
#pragma unroll
    for (int j = 0; j < 8; ++j) { r[j] = lo[j]; r[j + 8] = hi[j]; }
    return r;
}

// ---------------------------------------------------------------------------
// Kernel 1: per-head QKV projection. One wave computes a 16(t) x 64(s) tile of
// Q, K, V via 24 v_wmma_f32_16x16x32_f16. V is stored transposed [s][t] so the
// attention kernel's P@V B-fragments are contiguous.
// ---------------------------------------------------------------------------
__global__ void __launch_bounds__(128) qkv_kernel(
        const float* __restrict__ x,
        const float* __restrict__ Wq, const float* __restrict__ Wk,
        const float* __restrict__ Wv,
        _Float16* __restrict__ q16, _Float16* __restrict__ k16,
        _Float16* __restrict__ vT16)
{
    const int wid  = blockIdx.x * (blockDim.x >> 5) + (threadIdx.x >> 5); // 0..4095
    const int b    = wid >> 10;
    const int h    = (wid >> 6) & (H - 1);
    const int t0   = (wid & 63) << 4;
    const int bh   = b * H + h;
    const int lane = threadIdx.x & 31;
    const int nlo  = lane & 15;
    const int hi8  = (lane & 16) ? 8 : 0;
    const int koff = (lane & 16) ? 16 : 0;

    // A fragments: x rows t0..t0+15, head column slice, f32 -> f16
    const float* xr = x + ((size_t)(b * T + t0 + nlo)) * E + h * S;
    v16h a0, a1;
#pragma unroll
    for (int j = 0; j < 8; ++j) {
        a0[j]     = (_Float16)xr[koff + j];
        a0[j + 8] = (_Float16)xr[koff + 8 + j];
        a1[j]     = (_Float16)xr[32 + koff + j];
        a1[j + 8] = (_Float16)xr[32 + koff + 8 + j];
    }

    const float* Ws[3] = { Wq, Wk, Wv };
    for (int w = 0; w < 3; ++w) {
        v8f acc[4];
#pragma unroll
        for (int nc = 0; nc < 4; ++nc) acc[nc] = zero8f();
#pragma unroll
        for (int kc = 0; kc < 2; ++kc) {
            v16h a = kc ? a1 : a0;
#pragma unroll
            for (int nc = 0; nc < 4; ++nc) {
                // B[k][n] = W[n0+n][k]  (q = xh @ W^T)
                const float* wr = Ws[w] + (nc * 16 + nlo) * S + kc * 32 + koff;
                v16h bf;
#pragma unroll
                for (int j = 0; j < 16; ++j) bf[j] = (_Float16)wr[j];
                acc[nc] = WMMA_F16F32(a, bf, acc[nc]);
            }
        }
        if (w < 2) {
            _Float16* dst = (w == 0) ? q16 : k16;   // row-major [bh][t][s]
#pragma unroll
            for (int nc = 0; nc < 4; ++nc)
#pragma unroll
                for (int v = 0; v < 8; ++v)
                    dst[((size_t)bh * T + t0 + hi8 + v) * S + nc * 16 + nlo] =
                        (_Float16)acc[nc][v];
        } else {                                    // V transposed [bh][s][t]
#pragma unroll
            for (int nc = 0; nc < 4; ++nc) {
                union { _Float16 hv[8]; uint4 u; } pk;
#pragma unroll
                for (int v = 0; v < 8; ++v) pk.hv[v] = (_Float16)acc[nc][v];
                *(uint4*)(vT16 + ((size_t)bh * S + nc * 16 + nlo) * T + t0 + hi8) = pk.u;
            }
        }
    }
}

// ---------------------------------------------------------------------------
// Kernel 2: f32 -> f16 conversion of Er and Wo.
// ---------------------------------------------------------------------------
__global__ void cvt_kernel(const float* __restrict__ Er, const float* __restrict__ Wo,
                           _Float16* __restrict__ er16, _Float16* __restrict__ wo16)
{
    const int i  = blockIdx.x * blockDim.x + threadIdx.x;
    const int n1 = H * T * S;
    const int n2 = E * E;
    if (i < n1)            er16[i] = (_Float16)Er[i];
    else if (i < n1 + n2)  wo16[i - n1] = (_Float16)Wo[i - n1];
}

// ---------------------------------------------------------------------------
// Kernel 3: meanV[bh][s] = (1/T) * sum_t v[t][s]   (for padded query rows,
// where reference softmax is uniform over all T keys).
// ---------------------------------------------------------------------------
__global__ void meanv_kernel(const _Float16* __restrict__ vT16, float* __restrict__ meanv)
{
    const int i = blockIdx.x * blockDim.x + threadIdx.x;   // over B*H*S = 4096
    if (i >= B * H * S) return;
    const _Float16* p = vT16 + (size_t)i * T;
    float s = 0.0f;
    for (int t = 0; t < T; t += 8) {
        v8h vv = *(const v8h*)(p + t);
#pragma unroll
        for (int j = 0; j < 8; ++j) s += (float)vv[j];
    }
    meanv[i] = s * (1.0f / T);
}

// relative-position band tile from GLOBAL memory (used once for the initial
// "lo" band): 16 Er columns starting at row rc0 (clamped), K=64 -> 2 WMMAs
__device__ __forceinline__ v8f band_tile_g(const _Float16* __restrict__ erh, int rc0,
                                           v16h qa0, v16h qa1, int nlo, int koff)
{
    v8f acc = zero8f();
#pragma unroll
    for (int kc = 0; kc < 2; ++kc) {
        int r = rc0 + nlo;
        r = r < 0 ? 0 : (r > T - 1 ? T - 1 : r);
        v16h bf = *(const v16h*)(erh + (size_t)r * S + kc * 32 + koff);
        acc = WMMA_F16F32(kc ? qa1 : qa0, bf, acc);
    }
    return acc;
}

// ---------------------------------------------------------------------------
// Kernel 4: flash-style relative attention. One wave per (b,h,16 query rows).
// Per 32-column step: K rows and Er band rows (two contiguous 4KB blocks) are
// staged into LDS with double-buffered GLOBAL_LOAD_ASYNC_TO_LDS_B128 (ASYNCcnt)
// so WMMA work overlaps memory latency. 4 QK WMMAs + 4 band WMMAs (2 amortized
// via band reuse) + 4 P@V WMMAs; online softmax with shuffle reductions; P is
// staged through 1KB/wave LDS for C-layout -> A-fragment conversion.
// ---------------------------------------------------------------------------
__global__ void __launch_bounds__(128) attn_kernel(
        const _Float16* __restrict__ q16, const _Float16* __restrict__ k16,
        const _Float16* __restrict__ vT16, const _Float16* __restrict__ er16,
        const float* __restrict__ meanv, const int* __restrict__ maskp,
        _Float16* __restrict__ attn16)
{
    __shared__ _Float16 ldsP[4][16 * 32];       //  1KB/wave: P staging
    __shared__ _Float16 ldsK[4][2][32 * 64];    //  8KB/wave: K double buffer
    __shared__ _Float16 ldsE[4][2][32 * 64];    //  8KB/wave: Er band double buffer

    const int wslot = threadIdx.x >> 5;
    const int wid   = blockIdx.x * (blockDim.x >> 5) + wslot;
    const int b     = wid >> 10;
    const int h     = (wid >> 6) & (H - 1);
    const int i0    = (wid & 63) << 4;
    const int bh    = b * H + h;
    const int lane  = threadIdx.x & 31;
    const int nlo   = lane & 15;
    const int hi8   = (lane & 16) ? 8 : 0;
    const int koff  = (lane & 16) ? 16 : 0;
    _Float16* lp = ldsP[wslot];

    const _Float16* erh = er16 + (size_t)h * T * S;
    const _Float16* kb  = k16  + (size_t)bh * T * S;
    const _Float16* vb  = vT16 + (size_t)bh * S * T;

    // Q fragments, reused across all column steps
    const _Float16* qr = q16 + ((size_t)bh * T + i0 + nlo) * S;
    v16h qa0 = cat8(*(const v8h*)(qr + koff),      *(const v8h*)(qr + koff + 8));
    v16h qa1 = cat8(*(const v8h*)(qr + 32 + koff), *(const v8h*)(qr + 32 + koff + 8));

    v8f o[4];
#pragma unroll
    for (int nc = 0; nc < 4; ++nc) o[nc] = zero8f();
    float mrow[8], lrow[8];
#pragma unroll
    for (int v = 0; v < 8; ++v) { mrow[v] = -3.0e38f; lrow[v] = 0.0f; }

    // initial band chunk: Er rows [T-1-i0-16, T-1-i0-1] (one-time global read)
    v8f prevHi = band_tile_g(erh, T - 1 - i0 - 16, qa0, qa1, nlo, koff);

    // ---- async LDS staging ------------------------------------------------
    // Per step j0 we need K rows [j0, j0+31] (contiguous 4KB of k16) and Er
    // rows [T-1-i0+j0, +31] (contiguous 4KB of er16; base >= 15 always).
    // 8 lanes*16B chunks per 4KB block => 8 async b128 per block, 16 per step.
    const char* kgb = (const char*)kb;
    const char* egb = (const char*)erh;
    const long long gMax = (long long)(T * S * 2 - 16);   // clamp: masked cols anyway

    auto issue_tiles = [&](int j0n, int bufn) {
        const unsigned ldsKb = (unsigned)(size_t)&ldsK[wslot][bufn][0];
        const unsigned ldsEb = (unsigned)(size_t)&ldsE[wslot][bufn][0];
        const long long kB0 = (long long)j0n * (S * 2);
        const long long eB0 = (long long)(T - 1 - i0 + j0n) * (S * 2);
#pragma unroll
        for (int c = 0; c < 8; ++c) {
            const unsigned o = (unsigned)((c * 32 + lane) * 16);
            long long kByte = kB0 + o; if (kByte > gMax) kByte = gMax;
            long long eByte = eB0 + o; if (eByte > gMax) eByte = gMax;
            const unsigned long long gk = (unsigned long long)(kgb + kByte);
            const unsigned long long ge = (unsigned long long)(egb + eByte);
            asm volatile("global_load_async_to_lds_b128 %0, %1, off"
                         :: "v"(ldsKb + o), "v"(gk) : "memory");
            asm volatile("global_load_async_to_lds_b128 %0, %1, off"
                         :: "v"(ldsEb + o), "v"(ge) : "memory");
        }
    };

    issue_tiles(0, 0);
    int buf = 0;

    for (int j0 = 0; j0 <= i0; j0 += 32) {
        issue_tiles(j0 + 32, buf ^ 1);                 // prefetch next step
        asm volatile("s_wait_asynccnt 16" ::: "memory"); // current buffer ready

        const _Float16* lk = ldsK[wslot][buf];
        const _Float16* le = ldsE[wslot][buf];

        float s0[8], s1[8];
#pragma unroll
        for (int sub = 0; sub < 2; ++sub) {
            const int js = j0 + sub * 16;
            // QK^T tile (K B-fragments from LDS)
            v8f aqk = zero8f();
#pragma unroll
            for (int kc = 0; kc < 2; ++kc) {
                v16h bk = *(const v16h*)(lk + (sub * 16 + nlo) * S + kc * 32 + koff);
                aqk = WMMA_F16F32(kc ? qa1 : qa0, bk, aqk);
            }
            // relative-position band hi-chunk from LDS; reuse prev hi as lo
            v8f curHi = zero8f();
#pragma unroll
            for (int kc = 0; kc < 2; ++kc) {
                v16h bf = *(const v16h*)(le + (sub * 16 + nlo) * S + kc * 32 + koff);
                curHi = WMMA_F16F32(kc ? qa1 : qa0, bf, curHi);
            }
            v8f bl = prevHi;
            prevHi = curHi;
            float* sd = sub ? s1 : s0;
#pragma unroll
            for (int v = 0; v < 8; ++v) {
                const int ii  = v + hi8;                          // query row in tile
                const int src = ((lane - ii) & 15) | (lane & 16); // 16-lane rotate
                float lo = __shfl(bl[v], src, 32);
                float hh = __shfl(curHi[v], src, 32);
                float srel = (nlo >= ii) ? hh : lo;               // band col = jj-ii
                float sc = aqk[v] * QK_SCALE + srel;
                sd[v] = (js + nlo > i0 + ii) ? NEGV : sc;         // causal mask
            }
        }
        // joint online softmax over the 32 new columns
#pragma unroll
        for (int v = 0; v < 8; ++v) {
            float tm = fmaxf(s0[v], s1[v]);
#pragma unroll
            for (int d = 1; d < 16; d <<= 1) tm = fmaxf(tm, __shfl_xor(tm, d, 32));
            const float mn    = fmaxf(mrow[v], tm);
            const float alpha = __expf(mrow[v] - mn);
            mrow[v] = mn;
            const float p0 = __expf(s0[v] - mn);
            const float p1 = __expf(s1[v] - mn);
            float ps = p0 + p1;
#pragma unroll
            for (int d = 1; d < 16; d <<= 1) ps += __shfl_xor(ps, d, 32);
            lrow[v] = lrow[v] * alpha + ps;
#pragma unroll
            for (int nc = 0; nc < 4; ++nc) o[nc][v] *= alpha;
            const int ii = v + hi8;
            lp[ii * 32 + nlo]      = (_Float16)p0;
            lp[ii * 32 + 16 + nlo] = (_Float16)p1;
        }
        // P @ V : A fragment from LDS, B fragments from transposed V
        v16h pa = cat8(*(const v8h*)(lp + nlo * 32 + koff),
                       *(const v8h*)(lp + nlo * 32 + koff + 8));
#pragma unroll
        for (int nc = 0; nc < 4; ++nc) {
            int kst = j0 + koff; if (kst > T - 16) kst = T - 16; // clamped cols have p==0
            v16h bv = *(const v16h*)(vb + (size_t)(nc * 16 + nlo) * T + kst);
            o[nc] = WMMA_F16F32(pa, bv, o[nc]);
        }
        buf ^= 1;
    }

    // epilogue: normalize, substitute mean(V) for padded query rows, store
    // into attn16 laid out as [b][t][h*S+s] (ready for the output projection)
#pragma unroll
    for (int v = 0; v < 8; ++v) {
        const int t = i0 + v + hi8;
        const float inv = 1.0f / lrow[v];
        const bool pad = (maskp[b * T + t] == 0);
#pragma unroll
        for (int nc = 0; nc < 4; ++nc) {
            float val = o[nc][v] * inv;
            if (pad) val = meanv[bh * S + nc * 16 + nlo];
            attn16[((size_t)(b * T + t)) * E + h * S + nc * 16 + nlo] = (_Float16)val;
        }
    }
}

// ---------------------------------------------------------------------------
// Kernel 5: output projection  out = attn @ Wo^T + bo  (f16 WMMA, f32 out).
// One wave per 16x64 output tile; 128 WMMAs over K=1024.
// ---------------------------------------------------------------------------
__global__ void __launch_bounds__(128) proj_kernel(
        const _Float16* __restrict__ attn16, const _Float16* __restrict__ wo16,
        const float* __restrict__ bo, float* __restrict__ out)
{
    const int wid  = blockIdx.x * (blockDim.x >> 5) + (threadIdx.x >> 5); // 0..4095
    const int r0   = (wid >> 4) << 4;
    const int c0   = (wid & 15) << 6;
    const int lane = threadIdx.x & 31;
    const int nlo  = lane & 15;
    const int hi8  = (lane & 16) ? 8 : 0;
    const int koff = (lane & 16) ? 16 : 0;

    v8f acc[4];
#pragma unroll
    for (int nc = 0; nc < 4; ++nc) acc[nc] = zero8f();

    const _Float16* ar = attn16 + (size_t)(r0 + nlo) * E;
    for (int kc = 0; kc < E / 32; ++kc) {
        const _Float16* ap = ar + kc * 32;
        v16h a = cat8(*(const v8h*)(ap + koff), *(const v8h*)(ap + koff + 8));
#pragma unroll
        for (int nc = 0; nc < 4; ++nc) {
            // B[k][n] = Wo[c0+n][k]
            v16h bf = *(const v16h*)(wo16 + (size_t)(c0 + nc * 16 + nlo) * E + kc * 32 + koff);
            acc[nc] = WMMA_F16F32(a, bf, acc[nc]);
        }
    }
#pragma unroll
    for (int nc = 0; nc < 4; ++nc) {
        const float bias = bo[c0 + nc * 16 + nlo];
#pragma unroll
        for (int v = 0; v < 8; ++v)
            out[(size_t)(r0 + v + hi8) * E + c0 + nc * 16 + nlo] = acc[nc][v] + bias;
    }
}

// ---------------------------------------------------------------------------
extern "C" void kernel_launch(void* const* d_in, const int* in_sizes, int n_in,
                              void* d_out, int out_size, void* d_ws, size_t ws_size,
                              hipStream_t stream)
{
    (void)in_sizes; (void)n_in; (void)out_size; (void)ws_size;
    const float* x   = (const float*)d_in[0];
    const int*   msk = (const int*)d_in[1];
    const float* Wq  = (const float*)d_in[2];
    const float* Wk  = (const float*)d_in[3];
    const float* Wv  = (const float*)d_in[4];
    const float* Er  = (const float*)d_in[5];
    const float* Wo  = (const float*)d_in[6];
    const float* bo  = (const float*)d_in[7];
    float* out = (float*)d_out;

    char* ws = (char*)d_ws;
    const size_t MB = (size_t)1 << 20;
    _Float16* q16   = (_Float16*)(ws +  0 * MB);  //  8 MB  [B][H][T][S]
    _Float16* k16   = (_Float16*)(ws +  8 * MB);  //  8 MB  [B][H][T][S]
    _Float16* vT16  = (_Float16*)(ws + 16 * MB);  //  8 MB  [B][H][S][T]
    _Float16* er16  = (_Float16*)(ws + 24 * MB);  //  2 MB  [H][T][S]
    _Float16* wo16  = (_Float16*)(ws + 26 * MB);  //  2 MB  [E][E]
    _Float16* a16   = (_Float16*)(ws + 28 * MB);  //  8 MB  [B][T][E]
    float*    meanv = (float*)   (ws + 36 * MB);  // 16 KB  [B][H][S]

    qkv_kernel  <<<1024, 128, 0, stream>>>(x, Wq, Wk, Wv, q16, k16, vT16);
    cvt_kernel  <<<(2 * 1024 * 1024 + 255) / 256, 256, 0, stream>>>(Er, Wo, er16, wo16);
    meanv_kernel<<<16, 256, 0, stream>>>(vT16, meanv);
    attn_kernel <<<1024, 128, 0, stream>>>(q16, k16, vT16, er16, meanv, msk, a16);
    proj_kernel <<<1024, 128, 0, stream>>>(a16, wo16, bo, out);
}
